// Head_71476845740093
// MI455X (gfx1250) — matrix operand order repaired and checked by reading, not compiled, not run on hardware
//
#include <hip/hip_runtime.h>

typedef __attribute__((ext_vector_type(16))) __bf16 v16bf;
typedef __attribute__((ext_vector_type(8)))  __bf16 v8bf;
typedef __attribute__((ext_vector_type(8)))  float  v8f;
typedef __attribute__((ext_vector_type(4)))  unsigned int u32x4;
typedef __attribute__((ext_vector_type(8)))  unsigned int u32x8;

namespace {

constexpr int T  = 100;   // sequence length
constexpr int TP = 112;   // padded to 7*16
constexpr int C  = 16;    // embedding dim
constexpr int H  = 64;    // head size
constexpr int D  = 256;   // dense dim

constexpr int NWAVES   = 7;     // one 16-row strip per wave
constexpr int NTHREADS = NWAVES * 32;

// LDS pitches (bf16 elements). pitch*2 bytes is a multiple of 16 (aligned
// v8bf access) while the dword stride avoids 64-dword bank alignment.
constexpr int XP   = 40;   // x     [112][40]   K padded 16->32
constexpr int W1P  = 40;   // W1^T  [64][40]    K padded 16->32
constexpr int WDP  = 72;   // Wd^T  [256][72]
constexpr int T1P  = 72;   // t1    [112][72]
constexpr int QKP  = 264;  // q,k   [112][264]
constexpr int VTP  = 136;  // v^T   [256][136]  s padded 112->128
constexpr int WEIP = 136;  // wei   [112][136]  s padded 112->128

constexpr int SZ_X   = TP * XP  * 2;
constexpr int SZ_W1  = H  * W1P * 2;
constexpr int SZ_WD  = D  * WDP * 2;
constexpr int SZ_T1  = TP * T1P * 2;
constexpr int SZ_QK  = TP * QKP * 2;
constexpr int SZ_VT  = D  * VTP * 2;
constexpr int SZ_WEI = TP * WEIP * 2;

constexpr int OFF_X   = 0;
constexpr int OFF_W1  = OFF_X  + SZ_X;
constexpr int OFF_WD  = OFF_W1 + SZ_W1;
constexpr int OFF_T1  = OFF_WD + SZ_WD;
constexpr int OFF_Q   = OFF_T1 + SZ_T1;
constexpr int OFF_K   = OFF_Q  + SZ_QK;
constexpr int OFF_VT  = OFF_K  + SZ_QK;
constexpr int OFF_WEI = OFF_VT + SZ_VT;
constexpr int LDS_BYTES = OFF_WEI + SZ_WEI;   // 285,440 B < 320 KB

// f32 staging area for the TDM copy of x[b] (100x16 f32 = 6400 B);
// reuses the t1 region, which is written only after two later barriers.
constexpr int OFF_STAGE = OFF_T1;

// A operand (16x32): lane<16 holds row M=lane, K {0..7, 16..23};
// lane>=16 holds row M=lane-16, K {8..15, 24..31}. Row-major LDS source.
__device__ inline v16bf load_A(const __bf16* base, int pitch, int row0, int k0,
                               int lane) {
  int r = lane & 15, hi = lane >> 4;
  const __bf16* p = base + (row0 + r) * pitch + k0 + hi * 8;
  union { v16bf v; v8bf h[2]; } u;
  u.h[0] = *(const v8bf*)(p);
  u.h[1] = *(const v8bf*)(p + 16);
  return u.v;
}

// B operand (32x16): lane<16 holds col N=lane, K {0..7, 16..23}; lane>=16 K {8..15, 24..31}.
// Source is B^T stored row-major [N][K] -> two 16-byte chunks per lane.
__device__ inline v16bf load_B(const __bf16* bt, int pitch, int n0, int k0,
                               int lane) {
  int r = lane & 15, hi = lane >> 4;
  const __bf16* p = bt + (n0 + r) * pitch + k0 + hi * 8;
  union { v16bf v; v8bf h[2]; } u;
  u.h[0] = *(const v8bf*)(p);
  u.h[1] = *(const v8bf*)(p + 16);
  return u.v;
}

// D layout: lane<16 -> (M = 0..7, N = lane), lane>=16 -> (M = 8..15, N = lane-16)
__device__ inline void store_D_rowmajor(__bf16* base, int pitch, int row0,
                                        int n0, v8f d, int lane) {
  int r = lane & 15, hi = lane >> 4;
  v8bf pk = __builtin_convertvector(d, v8bf);   // packed v_cvt to bf16
  __bf16* p = base + (row0 + hi * 8) * pitch + n0 + r;
#pragma unroll
  for (int i = 0; i < 8; ++i) p[i * pitch] = pk[i];
}

__device__ inline void store_D_transposed(__bf16* vt, int pitch, int row0,
                                          int n0, v8f d, int lane) {
  int r = lane & 15, hi = lane >> 4;
  v8bf pk = __builtin_convertvector(d, v8bf);
  __bf16* p = vt + (n0 + r) * pitch + row0 + hi * 8;   // 16B aligned
  *(v8bf*)p = pk;
}

__device__ inline v8f wmma_bf16(v16bf a, v16bf b, v8f c) {
  return __builtin_amdgcn_wmma_f32_16x16x32_bf16(
      /*neg_a=*/false, a, /*neg_b=*/false, b,
      /*c_mod=*/(short)0, c, /*reuse_a=*/false, /*reuse_b=*/false);
}

}  // namespace

__global__ __launch_bounds__(NTHREADS, 1)
void head_attn_kernel(const float* __restrict__ x,
                      const float* __restrict__ Wk, const float* __restrict__ Wq,
                      const float* __restrict__ Wv, const float* __restrict__ Wdk,
                      const float* __restrict__ Wdq, const float* __restrict__ Wdv,
                      float* __restrict__ out) {
  extern __shared__ char smem[];
  __bf16* xs  = (__bf16*)(smem + OFF_X);
  __bf16* w1t = (__bf16*)(smem + OFF_W1);
  __bf16* wdt = (__bf16*)(smem + OFF_WD);
  __bf16* t1  = (__bf16*)(smem + OFF_T1);
  __bf16* qs  = (__bf16*)(smem + OFF_Q);
  __bf16* ks  = (__bf16*)(smem + OFF_K);
  __bf16* vt  = (__bf16*)(smem + OFF_VT);
  __bf16* wei = (__bf16*)(smem + OFF_WEI);

  const int b    = blockIdx.x;
  const int tid  = threadIdx.x;
  const int wave = tid >> 5;
  const int lane = tid & 31;
  const int row0 = wave * 16;          // this wave's t-strip

  const float* xb = x + (size_t)b * T * C;

  // L2 prefetch hints for the shared weights (reused by all 4096 blocks).
  {
    const float* ws[6] = {Wk, Wq, Wv, Wdk, Wdq, Wdv};
#pragma unroll
    for (int i = 0; i < 6; ++i) __builtin_prefetch(ws[i] + (tid << 4), 0, 0);
  }

  // TDM: async 2D tensor copy x[b] (100x16 f32) -> LDS staging (wave 0 only).
  if (wave == 0) {
    unsigned long long ga = (unsigned long long)(const void*)xb;
    u32x4 g0;
    g0[0] = 1u;                                   // count=1 (valid user D#)
    g0[1] = (unsigned)OFF_STAGE;                  // lds_addr (bytes)
    g0[2] = (unsigned)(ga & 0xFFFFFFFFu);         // global_addr[31:0]
    g0[3] = (unsigned)((ga >> 32) & 0x1FFFFFFu)   // global_addr[56:32]
          | 0x80000000u;                          // type=2 ("image")
    u32x8 g1;
    g1[0] = 2u << 16;                             // data_size = 4 bytes
    g1[1] = (unsigned)C << 16;                    // tensor_dim0[15:0] = 16
    g1[2] = (unsigned)T << 16;                    // tensor_dim1[15:0] = 100
    g1[3] = (unsigned)C << 16;                    // tile_dim0 = 16
    g1[4] = (unsigned)T;                          // tile_dim1 = 100
    g1[5] = (unsigned)C;                          // tensor_dim0_stride = 16
    g1[6] = 0u;
    g1[7] = 0u;
    asm volatile("tensor_load_to_lds %0, %1" :: "s"(g0), "s"(g1) : "memory");
    __builtin_amdgcn_s_wait_tensorcnt(0);
  }

  // Zero v^T and wei regions (their padded tails feed WMMA; stale NaN kills
  // it). 16-byte stores: both regions and the base offset are 16B aligned.
  {
    u32x4* z = (u32x4*)(smem + OFF_VT);
    const int n = (SZ_VT + SZ_WEI) >> 4;
    const u32x4 zero = {0u, 0u, 0u, 0u};
    for (int i = tid; i < n; i += NTHREADS) z[i] = zero;
  }
  __syncthreads();   // staging + zero regions visible

  // staging f32 -> bf16 LDS, zero-padded in K (16->32) and T (100->112)
  {
    const float* xstage = (const float*)(smem + OFF_STAGE);
    for (int e = tid; e < TP * 32; e += NTHREADS) {
      int r = e >> 5, c = e & 31;
      float v = (r < T && c < C) ? xstage[r * C + c] : 0.f;
      xs[r * XP + c] = (__bf16)v;
    }
  }
  __syncthreads();

  // Hoisted A operand: x strip tile (invariant across all 3 streams).
  const v16bf Ax = load_A(xs, XP, row0, 0, lane);

  const float* W1s[3] = {Wq, Wk, Wv};
  const float* Wds[3] = {Wdq, Wdk, Wdv};
  __bf16*      dst[3] = {qs, ks, nullptr};

#pragma unroll
  for (int s = 0; s < 3; ++s) {
    __syncthreads();  // protect shared W1t/Wdt/t1 reuse across streams
    const float* w1 = W1s[s];
    for (int e = tid; e < H * 32; e += NTHREADS) {         // W1^T [64][32]
      int n = e >> 5, k = e & 31;
      w1t[n * W1P + k] = (__bf16)(k < C ? w1[k * H + n] : 0.f);
    }
    const float* wd = Wds[s];
    for (int e = tid; e < D * H; e += NTHREADS) {          // Wd^T [256][64]
      int n = e >> 6, k = e & 63;
      wdt[n * WDP + k] = (__bf16)wd[k * D + n];
    }
    __syncthreads();

    // t1 strip = x_strip @ W1   (M=16, N=64, K=32 incl. zero pad)
#pragma unroll 2
    for (int nt = 0; nt < 4; ++nt) {
      v8f acc = {};
      acc = wmma_bf16(Ax, load_B(w1t, W1P, nt * 16, 0, lane), acc);
      store_D_rowmajor(t1, T1P, row0, nt * 16, acc, lane);
    }
    // Hoisted A operand: t1 strip tiles (invariant across the 16 N-tiles).
    v16bf At[2];
#pragma unroll
    for (int kc = 0; kc < 2; ++kc)
      At[kc] = load_A(t1, T1P, row0, kc * 32, lane);
    // stream strip = t1_strip @ Wd   (M=16, N=256, K=64)
#pragma unroll 2
    for (int nt = 0; nt < 16; ++nt) {
      v8f acc = {};
#pragma unroll
      for (int kc = 0; kc < 2; ++kc)
        acc = wmma_bf16(At[kc], load_B(wdt, WDP, nt * 16, kc * 32, lane), acc);
      if (s == 2) store_D_transposed(vt, VTP, row0, nt * 16, acc, lane);
      else        store_D_rowmajor(dst[s], QKP, row0, nt * 16, acc, lane);
    }
  }
  __syncthreads();  // q, k, v^T all resident

  // Hoisted A operand: q strip tiles (invariant across the 7 s-tiles).
  v16bf Aq[8];
#pragma unroll
  for (int kc = 0; kc < 8; ++kc)
    Aq[kc] = load_A(qs, QKP, row0, kc * 32, lane);

  // Scores: wei_strip = q_strip @ k^T, scale, causal mask (in registers)
  v8f sc[NWAVES];
#pragma unroll
  for (int st = 0; st < NWAVES; ++st) {
    v8f acc = {};
#pragma unroll
    for (int kc = 0; kc < 8; ++kc)
      acc = wmma_bf16(Aq[kc], load_B(ks, QKP, st * 16, kc * 32, lane), acc);
    const int scol = st * 16 + (lane & 15);
    const int trow = row0 + (lane >> 4) * 8;
#pragma unroll
    for (int i = 0; i < 8; ++i)
      sc[st][i] = (scol <= trow + i) ? acc[i] * 0.25f : -3.0e38f;  // C^-0.5
  }

  // Softmax in registers: half-wave (16-lane) butterfly reductions per row
  float mx[8], rs[8];
#pragma unroll
  for (int i = 0; i < 8; ++i) {
    float m = -3.0e38f;
#pragma unroll
    for (int st = 0; st < NWAVES; ++st) m = fmaxf(m, sc[st][i]);
#pragma unroll
    for (int off = 8; off; off >>= 1) m = fmaxf(m, __shfl_xor(m, off, 16));
    mx[i] = m;
  }
#pragma unroll
  for (int i = 0; i < 8; ++i) {
    float sum = 0.f;
#pragma unroll
    for (int st = 0; st < NWAVES; ++st) {
      float e = __expf(sc[st][i] - mx[i]);
      sc[st][i] = e;
      sum += e;
    }
#pragma unroll
    for (int off = 8; off; off >>= 1) sum += __shfl_xor(sum, off, 16);
    rs[i] = 1.0f / sum;
  }
  v8f rsv;
#pragma unroll
  for (int i = 0; i < 8; ++i) rsv[i] = rs[i];

  // wei strip -> bf16 LDS (private to this wave; no workgroup barrier needed)
  __bf16* wstrip = wei + wave * 16 * WEIP;
  {
    const int r = lane & 15, hi = lane >> 4;
#pragma unroll
    for (int st = 0; st < NWAVES; ++st) {
      v8bf pk = __builtin_convertvector(sc[st] * rsv, v8bf);  // packed cvt
#pragma unroll
      for (int i = 0; i < 8; ++i)
        wstrip[(hi * 8 + i) * WEIP + st * 16 + r] = pk[i];
    }
  }

  // Hoisted A operand: wei strip tiles (invariant across the 16 N-tiles).
  v16bf Aw[4];
#pragma unroll
  for (int kc = 0; kc < 4; ++kc)
    Aw[kc] = load_A(wstrip, WEIP, 0, kc * 32, lane);

  // out_strip = wei_strip @ v   (M=16, N=256, K=128 incl. zero pad)
  float* ob = out + (size_t)b * T * D;
#pragma unroll 2
  for (int nt = 0; nt < 16; ++nt) {
    v8f acc = {};
#pragma unroll
    for (int kc = 0; kc < 4; ++kc)
      acc = wmma_bf16(Aw[kc], load_B(vt, VTP, nt * 16, kc * 32, lane), acc);
    const int n = nt * 16 + (lane & 15);
    const int tb = row0 + (lane >> 4) * 8;
#pragma unroll
    for (int i = 0; i < 8; ++i)
      if (tb + i < T) ob[(size_t)(tb + i) * D + n] = acc[i];
  }
}

extern "C" void kernel_launch(void* const* d_in, const int* in_sizes, int n_in,
                              void* d_out, int out_size, void* d_ws,
                              size_t ws_size, hipStream_t stream) {
  (void)n_in; (void)out_size; (void)d_ws; (void)ws_size;
  const float* x   = (const float*)d_in[0];
  const float* Wk  = (const float*)d_in[1];
  const float* Wq  = (const float*)d_in[2];
  const float* Wv  = (const float*)d_in[3];
  const float* Wdk = (const float*)d_in[4];
  const float* Wdq = (const float*)d_in[5];
  const float* Wdv = (const float*)d_in[6];
  const int B = in_sizes[0] / (T * C);   // 4096
  head_attn_kernel<<<dim3(B), dim3(NTHREADS), LDS_BYTES, stream>>>(
      x, Wk, Wq, Wv, Wdk, Wdq, Wdv, (float*)d_out);
}
